// SelfAttention_80212809220580
// MI455X (gfx1250) — matrix operand order repaired and checked by reading, not compiled, not run on hardware
//
#include <hip/hip_runtime.h>
#include <math.h>

// Problem constants (match reference)
#define BB 4
#define SS 2048
#define CC 1024
#define HH 16
#define DD 64
#define NX (BB * SS * CC)  // 8388608 elements

typedef __attribute__((ext_vector_type(16))) _Float16 v16h;
typedef __attribute__((ext_vector_type(8)))  _Float16 v8h;
typedef __attribute__((ext_vector_type(8)))  float    v8f;

__device__ __forceinline__ v8f wmma32(v16h a, v16h b, v8f c) {
  // D = A(16x32 f16) * B(32x16 f16) + C(16x16 f32)
  return __builtin_amdgcn_wmma_f32_16x16x32_f16(false, a, false, b, (short)0, c,
                                                false, false);
}

// A-fragment loader: source row-major, row = M (or row of V^T), K contiguous.
// K order used here (contiguous, same as B): grp0 elems 0..15 -> K=k0..k0+15
//                        grp1 elems 0..15 -> K=k0+16..k0+31
__device__ __forceinline__ v16h load_a_frag(const _Float16* base, int ld,
                                            int row0, int k0) {
  const int lane = threadIdx.x & 31;
  const int m = lane & 15;
  const int grp = lane >> 4;
  const _Float16* p = base + (size_t)(row0 + m) * ld + k0 + 16 * grp;
  v8h lo = *(const v8h*)(p);
  v8h hi = *(const v8h*)(p + 8);
  return __builtin_shufflevector(lo, hi, 0, 1, 2, 3, 4, 5, 6, 7, 8, 9, 10, 11,
                                 12, 13, 14, 15);
}

// B-fragment loader: source holds column n as a contiguous-K row (e.g. W[o][c]).
// K order used here (contiguous, same as A): lane grp0 holds K=k0..k0+15, grp1 holds K=k0+16..k0+31.
__device__ __forceinline__ v16h load_b_frag(const _Float16* base, int ld,
                                            int col0, int k0) {
  const int lane = threadIdx.x & 31;
  const int n = lane & 15;
  const int grp = lane >> 4;
  return *(const v16h*)(base + (size_t)(col0 + n) * ld + k0 + 16 * grp);
}

// ---------------------------------------------------------------- pack f32->f16
__global__ void cvt_f32_f16(const float* __restrict__ in,
                            _Float16* __restrict__ out, int n) {
  int i = blockIdx.x * blockDim.x + threadIdx.x;
  if (i < n) out[i] = (_Float16)in[i];
}

// ---------------------------------------------------------------- QKV GEMM
// D[M=s, N=o] = sum_c x[s][c] * qkv_w[o][c]; epilogue: +bias, *D^-0.25 for q,k,
// scatter into per-head [B,H,S,D] f16 buffers. Each wave: 32x64 output tile
// (2 A-frags share 4 B-frags -> 8 WMMA per 12 b128 loads).
__global__ __launch_bounds__(256) void qkv_gemm(
    const _Float16* __restrict__ xh, const _Float16* __restrict__ wh,
    const float* __restrict__ bias, _Float16* __restrict__ qh,
    _Float16* __restrict__ kh, _Float16* __restrict__ vh) {
  const int wave = threadIdx.x >> 5;
  const int lane = threadIdx.x & 31;
  const int grp = lane >> 4;
  const int m0 = blockIdx.y * 256 + wave * 32;
  const int n0 = blockIdx.x * 64;

  const v8f zero = {0.f, 0.f, 0.f, 0.f, 0.f, 0.f, 0.f, 0.f};
  v8f acc[2][4];
#pragma unroll
  for (int mt = 0; mt < 2; ++mt)
#pragma unroll
    for (int nt = 0; nt < 4; ++nt) acc[mt][nt] = zero;

  for (int kk = 0; kk < CC; kk += 32) {
    if (kk + 128 < CC) {  // pull upcoming lines toward the WGP (global_prefetch_b8)
      __builtin_prefetch(xh + (size_t)(m0 + (lane & 15)) * CC + kk + 128, 0, 3);
      __builtin_prefetch(wh + (size_t)(n0 + lane) * CC + kk + 128, 0, 3);
    }
    v16h a0 = load_a_frag(xh, CC, m0, kk);
    v16h a1 = load_a_frag(xh, CC, m0 + 16, kk);
#pragma unroll
    for (int nt = 0; nt < 4; ++nt) {
      v16h b = load_b_frag(wh, CC, n0 + nt * 16, kk);
      acc[0][nt] = wmma32(a0, b, acc[0][nt]);
      acc[1][nt] = wmma32(a1, b, acc[1][nt]);
    }
  }

  const float inv4 = 0.35355339059327379f;  // 64^-0.25
#pragma unroll
  for (int nt = 0; nt < 4; ++nt) {
    int o = n0 + nt * 16 + (lane & 15);
    float bo = bias[o];
    int which = o >> 10;       // 0=q 1=k 2=v
    int within = o & 1023;
    int h = within >> 6;
    int d = within & 63;
    float scl = (which == 2) ? 1.0f : inv4;
    _Float16* dst = (which == 0) ? qh : (which == 1) ? kh : vh;
#pragma unroll
    for (int mt = 0; mt < 2; ++mt)
#pragma unroll
      for (int r = 0; r < 8; ++r) {
        int srow = m0 + mt * 16 + r + 8 * grp;  // flattened b*S+s
        int bidx = srow >> 11;                  // / S
        int sidx = srow & 2047;                 // % S
        dst[((size_t)(bidx * HH + h) * SS + sidx) * DD + d] =
            (_Float16)((acc[mt][nt][r] + bo) * scl);
      }
  }
}

// ---------------------------------------------------------------- attention
// One block = 4 waves = 64 query rows of one (b,h). Computes S^T = K*Q^T so
// softmax is lane-local (+ one shfl_xor(16)) and P converts to a B-fragment
// with shfl_xor(16) exchanges between lane halves. V staged transposed in LDS for the P*V gemm.
__global__ __launch_bounds__(128) void attn_fwd(const _Float16* __restrict__ qh,
                                                const _Float16* __restrict__ kh,
                                                const _Float16* __restrict__ vh,
                                                _Float16* __restrict__ aoh) {
  __shared__ __align__(16) _Float16 vT[DD * 32];  // [d][j] for a 32-key block

  const int bh = blockIdx.x;       // b*H + h
  const int tid = threadIdx.x;
  const int wave = tid >> 5;
  const int lane = tid & 31;
  const int grp = lane >> 4;
  const int s0 = blockIdx.y * 64 + wave * 16;

  const _Float16* qb = qh + (size_t)bh * SS * DD;
  const _Float16* kb = kh + (size_t)bh * SS * DD;
  const _Float16* vb = vh + (size_t)bh * SS * DD;

  // Q used as B-fragments of S^T = K*Q^T (lane = query column i)
  v16h qf0 = load_b_frag(qb, DD, s0, 0);
  v16h qf1 = load_b_frag(qb, DD, s0, 32);

  const v8f zero = {0.f, 0.f, 0.f, 0.f, 0.f, 0.f, 0.f, 0.f};
  v8f acc[4];  // out^T tiles: M = d-range t*16..t*16+15, N = query i
#pragma unroll
  for (int t = 0; t < 4; ++t) acc[t] = zero;
  float mrun = -1e30f, lrun = 0.f;

  const int jj = tid >> 2;           // 0..31: key row within block
  const int dchunk = (tid & 3) << 4; // 0,16,32,48

  for (int j0 = 0; j0 < SS; j0 += 32) {
    if (j0 + 32 < SS) {  // prefetch next block's K/V rows
      __builtin_prefetch(kb + (size_t)(j0 + 32 + jj) * DD + dchunk, 0, 3);
      __builtin_prefetch(vb + (size_t)(j0 + 32 + jj) * DD + dchunk, 0, 3);
    }
    __syncthreads();  // previous iteration's vT reads complete
    {  // cooperative stage: vT[d][j] = v[j0+j][d]
      const _Float16* vr = vb + (size_t)(j0 + jj) * DD + dchunk;
      v8h lo = *(const v8h*)vr;
      v8h hi = *(const v8h*)(vr + 8);
#pragma unroll
      for (int i = 0; i < 8; ++i) {
        vT[(dchunk + i) * 32 + jj] = lo[i];
        vT[(dchunk + 8 + i) * 32 + jj] = hi[i];
      }
    }
    __syncthreads();

    // S^T tiles: rows = keys j0..j0+15 and j0+16..j0+31, cols = 16 queries
    v16h ka0 = load_a_frag(kb, DD, j0, 0);
    v16h ka1 = load_a_frag(kb, DD, j0, 32);
    v16h kc0 = load_a_frag(kb, DD, j0 + 16, 0);
    v16h kc1 = load_a_frag(kb, DD, j0 + 16, 32);

    v8f st0 = zero;
    st0 = wmma32(ka0, qf0, st0);
    st0 = wmma32(ka1, qf1, st0);
    v8f st1 = zero;
    st1 = wmma32(kc0, qf0, st1);
    st1 = wmma32(kc1, qf1, st1);

    // online softmax over this 32-key block (per lane = one query column)
    float tmax = -1e30f;
#pragma unroll
    for (int r = 0; r < 8; ++r) tmax = fmaxf(tmax, fmaxf(st0[r], st1[r]));
    tmax = fmaxf(tmax, __shfl_xor(tmax, 16, 32));
    float mnew = fmaxf(mrun, tmax);
    float corr = __expf(mrun - mnew);
    float lsum = 0.f;

    // Build P as B-fragment for out^T = V^T * P:
    // grp0 needs K(j)=0..15, grp1 needs K(j)=16..31 -> one shfl_xor(16).
    v16h pf;
#pragma unroll
    for (int r = 0; r < 8; ++r) {
      float e0 = __expf(st0[r] - mnew);  // local rows j = r + 8*grp
      float e1 = __expf(st1[r] - mnew);  // local rows j = 16 + r + 8*grp
      lsum += e0 + e1;
      float x0 = __shfl_xor(e0, 16, 32);
      float x1 = __shfl_xor(e1, 16, 32);
      pf[r]     = (_Float16)(grp ? x1 : e0);
      pf[8 + r] = (_Float16)(grp ? e1 : x0);
    }
    lsum += __shfl_xor(lsum, 16, 32);
    lrun = lrun * corr + lsum;
    mrun = mnew;
#pragma unroll
    for (int t = 0; t < 4; ++t)
#pragma unroll
      for (int r = 0; r < 8; ++r) acc[t][r] *= corr;

    // V^T A-fragments from LDS (rows = d, K = j, ld = 32)
    v16h vf0 = load_a_frag(vT, 32, 0, 0);
    v16h vf1 = load_a_frag(vT, 32, 16, 0);
    v16h vf2 = load_a_frag(vT, 32, 32, 0);
    v16h vf3 = load_a_frag(vT, 32, 48, 0);
    acc[0] = wmma32(vf0, pf, acc[0]);
    acc[1] = wmma32(vf1, pf, acc[1]);
    acc[2] = wmma32(vf2, pf, acc[2]);
    acc[3] = wmma32(vf3, pf, acc[3]);
  }

  // epilogue: lane holds query i = lane&15; VGPR r of tile t is d = 16t+r+8*grp
  float invl = 1.0f / lrun;
  const int i = lane & 15;
  const int b = bh / HH, h = bh % HH;
  _Float16* dst = aoh + (((size_t)b * SS + (s0 + i)) * HH + h) * DD;
#pragma unroll
  for (int t = 0; t < 4; ++t)
#pragma unroll
    for (int r = 0; r < 8; ++r)
      dst[t * 16 + r + 8 * grp] = (_Float16)(acc[t][r] * invl);
}

// ---------------------------------------------------------------- unify GEMM
__global__ __launch_bounds__(256) void unify_gemm(
    const _Float16* __restrict__ ah, const _Float16* __restrict__ wh,
    const float* __restrict__ bias, float* __restrict__ out) {
  const int wave = threadIdx.x >> 5;
  const int lane = threadIdx.x & 31;
  const int grp = lane >> 4;
  const int m0 = blockIdx.y * 256 + wave * 32;
  const int n0 = blockIdx.x * 64;

  const v8f zero = {0.f, 0.f, 0.f, 0.f, 0.f, 0.f, 0.f, 0.f};
  v8f acc[2][4];
#pragma unroll
  for (int mt = 0; mt < 2; ++mt)
#pragma unroll
    for (int nt = 0; nt < 4; ++nt) acc[mt][nt] = zero;

  for (int kk = 0; kk < CC; kk += 32) {
    if (kk + 128 < CC) {
      __builtin_prefetch(ah + (size_t)(m0 + (lane & 15)) * CC + kk + 128, 0, 3);
      __builtin_prefetch(wh + (size_t)(n0 + lane) * CC + kk + 128, 0, 3);
    }
    v16h a0 = load_a_frag(ah, CC, m0, kk);
    v16h a1 = load_a_frag(ah, CC, m0 + 16, kk);
#pragma unroll
    for (int nt = 0; nt < 4; ++nt) {
      v16h b = load_b_frag(wh, CC, n0 + nt * 16, kk);
      acc[0][nt] = wmma32(a0, b, acc[0][nt]);
      acc[1][nt] = wmma32(a1, b, acc[1][nt]);
    }
  }

#pragma unroll
  for (int nt = 0; nt < 4; ++nt) {
    int o = n0 + nt * 16 + (lane & 15);
    float bo = bias[o];
#pragma unroll
    for (int mt = 0; mt < 2; ++mt)
#pragma unroll
      for (int r = 0; r < 8; ++r) {
        int srow = m0 + mt * 16 + r + 8 * grp;
        out[(size_t)srow * CC + o] = acc[mt][nt][r] + bo;
      }
  }
}

// ---------------------------------------------------------------- launch
extern "C" void kernel_launch(void* const* d_in, const int* in_sizes, int n_in,
                              void* d_out, int out_size, void* d_ws,
                              size_t ws_size, hipStream_t stream) {
  const float* x       = (const float*)d_in[0];
  const float* qkv_w   = (const float*)d_in[1];
  const float* qkv_b   = (const float*)d_in[2];
  const float* unify_w = (const float*)d_in[3];
  const float* unify_b = (const float*)d_in[4];
  float* out = (float*)d_out;

  // workspace layout (f16 elements), total ~92.3 MB
  _Float16* ws    = (_Float16*)d_ws;
  _Float16* xh    = ws;                        // NX
  _Float16* wqkvh = xh + NX;                   // 3*C*C
  _Float16* wuh   = wqkvh + 3 * CC * CC;       // C*C
  _Float16* qh    = wuh + CC * CC;             // NX  [B,H,S,D]
  _Float16* kh    = qh + NX;                   // NX
  _Float16* vh    = kh + NX;                   // NX
  _Float16* aoh   = vh + NX;                   // NX  [B,S,H,D]

  cvt_f32_f16<<<(NX + 255) / 256, 256, 0, stream>>>(x, xh, NX);
  cvt_f32_f16<<<(3 * CC * CC + 255) / 256, 256, 0, stream>>>(qkv_w, wqkvh,
                                                             3 * CC * CC);
  cvt_f32_f16<<<(CC * CC + 255) / 256, 256, 0, stream>>>(unify_w, wuh, CC * CC);

  // M=8192 rows in 256-row blocks; N in 64-col blocks
  qkv_gemm<<<dim3((3 * CC) / 64, (BB * SS) / 256), 256, 0, stream>>>(
      xh, wqkvh, qkv_b, qh, kh, vh);

  attn_fwd<<<dim3(BB * HH, SS / 64), 128, 0, stream>>>(qh, kh, vh, aoh);

  unify_gemm<<<dim3(CC / 64, (BB * SS) / 256), 256, 0, stream>>>(aoh, wuh,
                                                                 unify_b, out);
}
